// SNNModel_63745904608018
// MI455X (gfx1250) — matrix operand order repaired and checked by reading, not compiled, not run on hardware
//
#include <hip/hip_runtime.h>

// ---------------------------------------------------------------------------
// SNN forward on MI455X (gfx1250):
//   Z1 = X^T * W1^T   (bf16 WMMA GEMM, fp32 accum)   [4096 x 8192]
//   scan1: v1 = v1*0.9 + Z1[t]; s1 = v1>=1; reset    -> S1 (bf16, {0,1})
//   Z2 = S1 * W2^T    (bf16 WMMA GEMM, fp32 accum)   [4096 x 2048]
//   scan2: v2 likewise                               -> S2
//   transpose S1,S2 (time-major) -> d_out ([neuron][t], fp32)
//
// CDNA5 paths used: v_wmma_f32_16x16x32_bf16 (matrix core),
// global_load_async_to_lds_b128 + s_wait_asynccnt (async A-tile staging).
// ---------------------------------------------------------------------------

typedef __attribute__((ext_vector_type(16))) __bf16 v16bf;
typedef __attribute__((ext_vector_type(8)))  __bf16 v8bf;
typedef __attribute__((ext_vector_type(4)))  __bf16 v4bf;
typedef __attribute__((ext_vector_type(8)))  float  v8f;

#define N_IN    2048
#define N_HID   8192
#define N_OUT   2048
#define T_FULL  4096
#define T_STEPS 4095   // loop starts at t=1 in the reference
#define T_PAD   4096   // padded time dim for GEMM tiling (last col = 0)

#define DECAY 0.9f
#define VTH   1.0f

// ---------------------------------------------------------------------------
// GEMM: C[M x N] (fp32, row-major) = A[M x K] (bf16, row-major) *
//                                    B[K x N] (bf16, row-major)
// Requires M%128==0, N%128==0, K%32==0 (true for all call sites).
// Block = 256 threads = 8 waves; block tile 128x128, K-step 32.
// Wave grid 4x2; each wave computes 32x64 = 2x4 WMMA 16x16 tiles.
// A tile staged via per-lane async global->LDS DMA (ASYNCcnt);
// B tile staged through VGPRs because it is transposed on the LDS store.
// ---------------------------------------------------------------------------
__global__ __launch_bounds__(256)
void gemm_bf16_wmma(const __bf16* __restrict__ A, const __bf16* __restrict__ B,
                    float* __restrict__ C, int M, int N, int K)
{
    constexpr int BM = 128, BN = 128, BK = 32, PAD = 8; // 80B row stride: 16B-aligned, bank-staggered
    __shared__ __attribute__((aligned(16))) __bf16 As[BM][BK + PAD];
    __shared__ __attribute__((aligned(16))) __bf16 Bs[BN][BK + PAD]; // transposed: Bs[n][k]

    const int tid  = threadIdx.x;
    const int lane = tid & 31;
    const int w    = tid >> 5;
    const int wr   = w >> 1;        // 0..3 : wave row   (32 rows each)
    const int wc   = w & 1;         // 0..1 : wave col   (64 cols each)
    const int half = lane >> 4;     // 0/1  : lane half (ISA K-group select)
    const int lm   = lane & 15;
    const int mBase = blockIdx.y * BM;
    const int nBase = blockIdx.x * BN;

    v8f acc[8];
    #pragma unroll
    for (int i = 0; i < 8; ++i)
        #pragma unroll
        for (int e = 0; e < 8; ++e) acc[i][e] = 0.0f;

    for (int kt = 0; kt < K; kt += BK) {
        // ---- A tile: 128x32 bf16 = 8 KB, straight copy global->LDS.
        // 512 chunks of 16B; each lane DMAs 2 chunks via async-to-LDS.
        #pragma unroll
        for (int it = 0; it < 2; ++it) {
            int c   = it * 256 + tid;          // 0..511
            int row = c >> 2;                  // 4 chunks per 32-wide row
            int kk  = (c & 3) * 8;             // 8 bf16 = 16 bytes
            const __bf16* gptr = A + (size_t)(mBase + row) * K + kt + kk;
            unsigned loff = (unsigned)(unsigned long long)(&As[row][kk]); // LDS byte addr = low 32 bits
            asm volatile("global_load_async_to_lds_b128 %0, %1, off"
                         :: "v"(loff), "v"(gptr)
                         : "memory");
        }
        // ---- B tile: 32x128 bf16, stored transposed Bs[n][k] (VGPR path).
        #pragma unroll
        for (int it = 0; it < 4; ++it) {
            int c  = it * 256 + tid;
            int kr = c >> 5;               // 32 chunks per 128-wide row
            int nn = (c & 31) * 4;
            v4bf vb = *(const v4bf*)(B + (size_t)(kt + kr) * N + nBase + nn);
            Bs[nn + 0][kr] = vb[0];
            Bs[nn + 1][kr] = vb[1];
            Bs[nn + 2][kr] = vb[2];
            Bs[nn + 3][kr] = vb[3];
        }
        // Async A-tile writes must land before the barrier opens compute.
        asm volatile("s_wait_asynccnt 0x0" ::: "memory");
        __syncthreads();

        // ---- Build fragments per ISA VGPR layouts (wave32).
        // A 16x32 bf16: lane half 0 -> K {0..7, 16..23}; half 1 -> K {8..15, 24..31}
        v16bf afr[2];
        #pragma unroll
        for (int i = 0; i < 2; ++i) {
            int m = wr * 32 + i * 16 + lm;
            v8bf lo = *(const v8bf*)(&As[m][8 * half]);
            v8bf hi = *(const v8bf*)(&As[m][16 + 8 * half]);
            afr[i] = __builtin_shufflevector(lo, hi, 0,1,2,3,4,5,6,7,8,9,10,11,12,13,14,15);
        }
        // B 32x16 bf16: lane half 0 -> K {0..7, 16..23}; half 1 -> K {8..15, 24..31}
        v16bf bfr[4];
        #pragma unroll
        for (int j = 0; j < 4; ++j) {
            int n = wc * 64 + j * 16 + lm;
            v8bf lo = *(const v8bf*)(&Bs[n][8 * half]);
            v8bf hi = *(const v8bf*)(&Bs[n][16 + 8 * half]);
            bfr[j] = __builtin_shufflevector(lo, hi, 0,1,2,3,4,5,6,7,8,9,10,11,12,13,14,15);
        }

        #pragma unroll
        for (int i = 0; i < 2; ++i)
            #pragma unroll
            for (int j = 0; j < 4; ++j)
                acc[i * 4 + j] = __builtin_amdgcn_wmma_f32_16x16x32_bf16(
                    /*neg_a=*/false, afr[i], /*neg_b=*/false, bfr[j],
                    /*c_mod=*/(short)0, acc[i * 4 + j],
                    /*reuse_a=*/false, /*reuse_b=*/false);

        __syncthreads();  // protect LDS from next iteration's async writes
    }

    // ---- Epilogue per 32-bit C/D layout: VGPR r -> M = 8*half + r, N = lane&15
    #pragma unroll
    for (int i = 0; i < 2; ++i)
        #pragma unroll
        for (int j = 0; j < 4; ++j)
            #pragma unroll
            for (int r = 0; r < 8; ++r) {
                int m = mBase + wr * 32 + i * 16 + half * 8 + r;
                int n = nBase + wc * 64 + j * 16 + lm;
                C[(size_t)m * N + n] = acc[i * 4 + j][r];
            }
}

// ---------------------------------------------------------------------------
// XhT[t][k] = stim[0][k][t+1] (bf16), t in [0,4095); pad row t=4095 -> 0
// ---------------------------------------------------------------------------
__global__ void build_xht(const float* __restrict__ stim, __bf16* __restrict__ XhT)
{
    int k = blockIdx.x * 16 + threadIdx.x;   // 0..2047
    int t = blockIdx.y * 16 + threadIdx.y;   // 0..4095
    float v = (t < T_STEPS) ? stim[(size_t)k * T_FULL + t + 1] : 0.0f;
    XhT[(size_t)t * N_IN + k] = (__bf16)v;
}

// dst[c][r] (bf16, [cols x rows]) = src[r][c] (fp32, [rows x cols])
__global__ void transpose_f32_to_bf16(const float* __restrict__ src, __bf16* __restrict__ dst,
                                      int rows, int cols)
{
    int c = blockIdx.x * 16 + threadIdx.x;
    int r = blockIdx.y * 16 + threadIdx.y;
    if (r < rows && c < cols)
        dst[(size_t)c * rows + r] = (__bf16)src[(size_t)r * cols + c];
}

// ---------------------------------------------------------------------------
// Leaky-integrate / threshold / reset scan. One thread per neuron, coalesced
// time-major reads. fp32 state, exact reference semantics.
// ---------------------------------------------------------------------------
__global__ void snn_scan(const float* __restrict__ ZT, __bf16* __restrict__ ShT, int NN)
{
    int n = blockIdx.x * blockDim.x + threadIdx.x;
    float v = 0.0f;
    for (int t = 0; t < T_STEPS; ++t) {
        v = v * DECAY + ZT[(size_t)t * NN + n];
        float s = (v >= VTH) ? 1.0f : 0.0f;
        v = (v >= VTH) ? 0.0f : v;           // v *= (1 - s)
        ShT[(size_t)t * NN + n] = (__bf16)s; // {0,1} exact in bf16
    }
    ShT[(size_t)T_STEPS * NN + n] = (__bf16)0.0f; // pad column for next GEMM
}

// out[n][t] (fp32, [NN x 4095]) = ShT[t][n]
__global__ void emit_spikes(const __bf16* __restrict__ ShT, float* __restrict__ out, int NN)
{
    int n = blockIdx.x * 16 + threadIdx.x;
    int t = blockIdx.y * 16 + threadIdx.y;
    if (t < T_STEPS)
        out[(size_t)n * T_STEPS + t] = (float)ShT[(size_t)t * NN + n];
}

// ---------------------------------------------------------------------------
extern "C" void kernel_launch(void* const* d_in, const int* in_sizes, int n_in,
                              void* d_out, int out_size, void* d_ws, size_t ws_size,
                              hipStream_t stream)
{
    const float* stim = (const float*)d_in[0]; // [1, 2048, 4096]
    const float* W1   = (const float*)d_in[1]; // [8192, 2048]
    const float* W2   = (const float*)d_in[2]; // [2048, 8192]
    float* out = (float*)d_out;                // S1 [8192x4095] ++ S2 [2048x4095]

    // Workspace layout (272 MB), with temporally-disjoint buffers aliased:
    char* ws = (char*)d_ws;
    size_t off = 0;
    __bf16* XhT  = (__bf16*)(ws + off); off += (size_t)T_PAD * N_IN  * 2; // 16 MB (later: S2hT)
    __bf16* W1T  = (__bf16*)(ws + off); off += (size_t)N_IN  * N_HID * 2; // 32 MB (later: Z2T)
    __bf16* W2T  = (__bf16*)(ws + off); off += (size_t)N_HID * N_OUT * 2; // 32 MB
    __bf16* S1hT = (__bf16*)(ws + off); off += (size_t)T_PAD * N_HID * 2; // 64 MB
    float*  Z1T  = (float*) (ws + off); off += (size_t)T_PAD * N_HID * 4; // 128 MB
    __bf16* S2hT = XhT;          // XhT dead after GEMM1
    float*  Z2T  = (float*)W1T;  // W1T dead after GEMM1

    dim3 thr16(16, 16);

    // Precision conversion / transposes (bf16 inputs for WMMA, time-major X)
    build_xht<<<dim3(N_IN / 16, T_PAD / 16), thr16, 0, stream>>>(stim, XhT);
    transpose_f32_to_bf16<<<dim3(N_IN / 16,  N_HID / 16), thr16, 0, stream>>>(W1, W1T, N_HID, N_IN);
    transpose_f32_to_bf16<<<dim3(N_HID / 16, N_OUT / 16), thr16, 0, stream>>>(W2, W2T, N_OUT, N_HID);

    // Layer 1: Z1T[t][h] = sum_k x_t[k] * W1[h][k]
    gemm_bf16_wmma<<<dim3(N_HID / 128, T_PAD / 128), 256, 0, stream>>>(
        XhT, W1T, Z1T, T_PAD, N_HID, N_IN);
    snn_scan<<<N_HID / 256, 256, 0, stream>>>(Z1T, S1hT, N_HID);

    // Layer 2: Z2T[t][o] = sum_h s1_t[h] * W2[o][h]
    gemm_bf16_wmma<<<dim3(N_OUT / 128, T_PAD / 128), 256, 0, stream>>>(
        S1hT, W2T, Z2T, T_PAD, N_OUT, N_HID);
    snn_scan<<<N_OUT / 256, 256, 0, stream>>>(Z2T, S2hT, N_OUT);

    // Emit reference layout [neuron][t]
    emit_spikes<<<dim3(N_HID / 16, (T_STEPS + 15) / 16), thr16, 0, stream>>>(S1hT, out, N_HID);
    emit_spikes<<<dim3(N_OUT / 16, (T_STEPS + 15) / 16), thr16, 0, stream>>>(
        S2hT, out + (size_t)N_HID * T_STEPS, N_OUT);
}
